// semidual_15375982920139
// MI455X (gfx1250) — matrix operand order repaired and checked by reading, not compiled, not run on hardware
//
#include <hip/hip_runtime.h>
#include <hip/hip_bf16.h>

#define N_ROWS 16384
#define M_COLS 16384
#define DIM    256
#define WAVES  8       // waves per block in GEMM kernel (128 rows/block)
#define CSPLIT 8       // column-space split across blockIdx.y

#define LDS_STRIDE 264                    // 256 + 8 bf16 pad: conflict-free ds_load pattern
#define SBUF (2 * 16 * LDS_STRIDE)        // one double-buffer slot (hi+lo, 16 rows)

typedef __attribute__((ext_vector_type(16))) __bf16 v16bf;
typedef __attribute__((ext_vector_type(8)))  __bf16 v8bf;
typedef __attribute__((ext_vector_type(8)))  float  v8f;
typedef int v4i __attribute__((vector_size(16)));   // b128 payload type for async-LDS builtin

#if defined(__HIP_DEVICE_COMPILE__) && defined(__gfx1250__) && \
    __has_builtin(__builtin_amdgcn_global_load_async_to_lds_b128) && \
    __has_builtin(__builtin_amdgcn_s_wait_asynccnt)
#define USE_ASYNC_LDS 1
#else
#define USE_ASYNC_LDS 0
#endif

// Order-independent (deterministic) float atomic min via int/uint monotone trick.
__device__ __forceinline__ void atomic_min_float(float* addr, float val) {
    if (val >= 0.0f) atomicMin((int*)addr, __float_as_int(val));
    else             atomicMax((unsigned int*)addr, __float_as_uint(val));
}

// One wave per row: squared norm (optionally minus psi); optionally init rowmin to +inf.
__global__ void prep_rows_kernel(const float* __restrict__ src,
                                 const float* __restrict__ psi,
                                 float* __restrict__ outv,
                                 float* __restrict__ rowmin) {
    const int wid  = threadIdx.x >> 5;
    const int lane = threadIdx.x & 31;
    const int row  = blockIdx.x * (blockDim.x >> 5) + wid;
    const float* p = src + (size_t)row * DIM;
    float s = 0.0f;
    #pragma unroll
    for (int t = 0; t < DIM; t += 32) {
        float v = p[t + lane];
        s = fmaf(v, v, s);
    }
    #pragma unroll
    for (int off = 16; off > 0; off >>= 1) s += __shfl_xor(s, off, 32);
    if (lane == 0) {
        outv[row] = psi ? (s - psi[row]) : s;
        if (rowmin) rowmin[row] = __int_as_float(0x7F800000);  // +inf
    }
}

// Elementwise fp32 -> bf16 hi/lo split (x = hi + lo).
__global__ void convert_kernel(const float* __restrict__ src,
                               __bf16* __restrict__ hi,
                               __bf16* __restrict__ lo, int count) {
    int i = blockIdx.x * blockDim.x + threadIdx.x;
    if (i < count) {
        float v  = src[i];
        __bf16 h = (__bf16)v;
        hi[i] = h;
        lo[i] = (__bf16)(v - (float)h);
    }
}

// Cooperative stage of one 16-column y tile (hi+lo) into an LDS buffer.
// 1024 chunks of 16B; 256 threads copy 4 chunks each (async global->LDS).
__device__ __forceinline__ void stage_tile(const __bf16* __restrict__ yh,
                                           const __bf16* __restrict__ yl,
                                           int j0, __bf16* sdst) {
    const int t = threadIdx.x;  // 0..255
    #pragma unroll
    for (int q = 0; q < 4; ++q) {
        const int c   = q * 256 + t;       // 0..1023
        const int h   = c >> 9;            // 0: hi array, 1: lo array
        const int c2  = c & 511;
        const int row = c2 >> 5;           // 32 x 16B chunks per row
        const int cc  = c2 & 31;
        const __bf16* g = (h ? yl : yh) + (size_t)(j0 + row) * DIM + cc * 8;
        __bf16* l = sdst + (h * 16 + row) * LDS_STRIDE + cc * 8;
#if USE_ASYNC_LDS
        __builtin_amdgcn_global_load_async_to_lds_b128(
            (v4i*)g,
            (__attribute__((address_space(3))) v4i*)l, 0, 0);
#else
        *(v8bf*)l = *(const v8bf*)g;
#endif
    }
}

// Load one B fragment (16 contiguous bf16 as two 16B-aligned halves) from LDS.
__device__ __forceinline__ v16bf ldb(const __bf16* base, int kc) {
    v8bf a = *(const v8bf*)(base + kc * 32);
    v8bf b = *(const v8bf*)(base + kc * 32 + 8);
    return __builtin_shufflevector(a, b, 0,1,2,3,4,5,6,7,8,9,10,11,12,13,14,15);
}

// Fused bf16x3 GEMM + min epilogue.
// Block: 8 waves, 128 rows; y tiles staged through double-buffered LDS (async TDM-style).
__global__ void __launch_bounds__(WAVES * 32)
__attribute__((amdgpu_waves_per_eu(1)))
gemm_min_kernel(const __bf16* __restrict__ xh, const __bf16* __restrict__ xl,
                const __bf16* __restrict__ yh, const __bf16* __restrict__ yl,
                const float*  __restrict__ x2, const float*  __restrict__ w,
                float* __restrict__ rowmin) {
    __shared__ __bf16 sy[2][SBUF];

    const int lane = threadIdx.x & 31;
    const int wid  = threadIdx.x >> 5;
    const int m16  = lane & 15;   // A row / B col / C col within tile
    const int kh   = lane >> 4;   // lane half selects K sub-range
    const int r0   = (blockIdx.x * WAVES + wid) * 16;

    // A fragments (hi & lo) for full K=256, resident in VGPRs for the whole scan.
    v16bf ahi[8], alo[8];
    {
        const __bf16* xr_h = xh + (size_t)(r0 + m16) * DIM;
        const __bf16* xr_l = xl + (size_t)(r0 + m16) * DIM;
        #pragma unroll
        for (int kc = 0; kc < 8; ++kc) {
            const int kb = kc * 32 + kh * 16;  // contiguous K layout: 16 k per lane half, matches B
            v8bf h0 = *(const v8bf*)(xr_h + kb);
            v8bf h1 = *(const v8bf*)(xr_h + kb + 8);
            v8bf l0 = *(const v8bf*)(xr_l + kb);
            v8bf l1 = *(const v8bf*)(xr_l + kb + 8);
            ahi[kc] = __builtin_shufflevector(h0, h1, 0,1,2,3,4,5,6,7,8,9,10,11,12,13,14,15);
            alo[kc] = __builtin_shufflevector(l0, l1, 0,1,2,3,4,5,6,7,8,9,10,11,12,13,14,15);
        }
    }

    float minv[8];
    #pragma unroll
    for (int r = 0; r < 8; ++r) minv[r] = __int_as_float(0x7F800000);

    const int jBeg = blockIdx.y * (M_COLS / CSPLIT);
    const int jEnd = jBeg + (M_COLS / CSPLIT);

    stage_tile(yh, yl, jBeg, sy[0]);

    int it = 0;
    for (int j0 = jBeg; j0 < jEnd; j0 += 16, ++it) {
        const int p = it & 1;
#if USE_ASYNC_LDS
        __builtin_amdgcn_s_wait_asynccnt(0);
#endif
        __syncthreads();                              // buf[p] ready for all waves
        if (j0 + 16 < jEnd) stage_tile(yh, yl, j0 + 16, sy[p ^ 1]);  // overlap with compute

        const __bf16* sh_ = sy[p] + (0 * 16 + m16) * LDS_STRIDE + kh * 16;
        const __bf16* sl_ = sy[p] + (1 * 16 + m16) * LDS_STRIDE + kh * 16;

        // Entire lo-B half of the tile register-resident: one burst of 16 ds_loads.
        // The empty asm pins all 8 fragments live here so the loads cannot be sunk
        // to their uses (which re-serializes them behind s_wait_dscnt 0).
        v16bf blr[8];
        #pragma unroll
        for (int kc = 0; kc < 8; ++kc) blr[kc] = ldb(sl_, kc);
        #pragma unroll
        for (int kc = 0; kc < 8; ++kc) asm volatile("" : "+v"(blr[kc]));

        v8f acc = {};
        v16bf bh = ldb(sh_, 0);
        #pragma unroll
        for (int kc = 0; kc < 8; ++kc) {
            v16bf bhn = bh;
            if (kc < 7) bhn = ldb(sh_, kc + 1);       // 1 chunk ahead
            acc = __builtin_amdgcn_wmma_f32_16x16x32_bf16(false, ahi[kc], false, bh,
                                                          (short)0, acc, false, false);
            acc = __builtin_amdgcn_wmma_f32_16x16x32_bf16(false, alo[kc], false, bh,
                                                          (short)0, acc, false, false);
            acc = __builtin_amdgcn_wmma_f32_16x16x32_bf16(false, ahi[kc], false, blr[kc],
                                                          (short)0, acc, false, false);
            bh = bhn;
        }

        const float wv = w[j0 + m16];
        #pragma unroll
        for (int r = 0; r < 8; ++r) {
            // track min_j (w_j - 2<x,y>); x2 (constant per row) added in epilogue
            minv[r] = fminf(minv[r], fmaf(-2.0f, acc[r], wv));
        }
        __syncthreads();                              // all waves done reading buf[p]
    }

    // Reduce each row's 16 column-lanes (masks <16 stay within each 16-lane half).
    #pragma unroll
    for (int r = 0; r < 8; ++r) {
        float v = minv[r];
        v = fminf(v, __shfl_xor(v, 8, 32));
        v = fminf(v, __shfl_xor(v, 4, 32));
        v = fminf(v, __shfl_xor(v, 2, 32));
        v = fminf(v, __shfl_xor(v, 1, 32));
        if (m16 == 0) {
            const int row = r0 + kh * 8 + r;          // C row = r + 8*(lane/16)
            atomic_min_float(&rowmin[row], v + x2[row]);
        }
    }
}

// Single-block deterministic reduction -> (mean rowmin, mean psi).
__global__ void finalize_kernel(const float* __restrict__ rowmin,
                                const float* __restrict__ psi,
                                float* __restrict__ out) {
    __shared__ float s1[256], s2[256];
    const int t = threadIdx.x;
    float a = 0.0f, b = 0.0f;
    for (int i = t; i < N_ROWS; i += 256) a += rowmin[i];
    for (int i = t; i < M_COLS; i += 256) b += psi[i];
    s1[t] = a; s2[t] = b;
    __syncthreads();
    for (int off = 128; off > 0; off >>= 1) {
        if (t < off) { s1[t] += s1[t + off]; s2[t] += s2[t + off]; }
        __syncthreads();
    }
    if (t == 0) {
        out[0] = s1[0] / (float)N_ROWS;
        out[1] = s2[0] / (float)M_COLS;
    }
}

extern "C" void kernel_launch(void* const* d_in, const int* in_sizes, int n_in,
                              void* d_out, int out_size, void* d_ws, size_t ws_size,
                              hipStream_t stream) {
    const float* x   = (const float*)d_in[0];
    const float* y   = (const float*)d_in[1];
    const float* psi = (const float*)d_in[2];
    float* out = (float*)d_out;

    char* ws = (char*)d_ws;
    size_t off = 0;
    __bf16* xh = (__bf16*)(ws + off); off += (size_t)N_ROWS * DIM * 2;
    __bf16* xl = (__bf16*)(ws + off); off += (size_t)N_ROWS * DIM * 2;
    __bf16* yh = (__bf16*)(ws + off); off += (size_t)M_COLS * DIM * 2;
    __bf16* yl = (__bf16*)(ws + off); off += (size_t)M_COLS * DIM * 2;
    float* x2     = (float*)(ws + off); off += (size_t)N_ROWS * 4;
    float* w      = (float*)(ws + off); off += (size_t)M_COLS * 4;
    float* rowmin = (float*)(ws + off); off += (size_t)N_ROWS * 4;

    convert_kernel<<<(N_ROWS * DIM) / 256, 256, 0, stream>>>(x, xh, xl, N_ROWS * DIM);
    convert_kernel<<<(M_COLS * DIM) / 256, 256, 0, stream>>>(y, yh, yl, M_COLS * DIM);
    prep_rows_kernel<<<N_ROWS / 8, 256, 0, stream>>>(x, nullptr, x2, rowmin);
    prep_rows_kernel<<<M_COLS / 8, 256, 0, stream>>>(y, psi, w, nullptr);

    dim3 grid(N_ROWS / (16 * WAVES), CSPLIT);
    gemm_min_kernel<<<grid, WAVES * 32, 0, stream>>>(xh, xl, yh, yl, x2, w, rowmin);

    finalize_kernel<<<1, 256, 0, stream>>>(rowmin, psi, out);
}
